// RNNModel_7327214207080
// MI455X (gfx1250) — compile-verified, matrix operand order checked by reading, not run on hardware
//
#include <hip/hip_runtime.h>
#include <hip/hip_bf16.h>

// Problem constants (match reference)
#define T_  50
#define B_  64
#define E_  1024
#define H_  1024
#define V_  32000
#define H4_ (4 * H_)

typedef __attribute__((ext_vector_type(16))) _Float16 v16h;
typedef __attribute__((ext_vector_type(8)))  _Float16 v8h;
typedef __attribute__((ext_vector_type(8)))  float    v8f;

// ---------------------------------------------------------------------------
// WMMA helpers (CDNA5, wave32). f16 inputs, f32 accumulate.
// Fragment layout per CDNA5 ISA §7.12.2 (16-bit A 16x32):
//   lane<16 : row = lane,     K = {k0..k0+7, k0+16..k0+23}
//   lane>=16: row = lane-16,  K = {k0+8..k0+15, k0+24..k0+31}
// Same layout used for B loaded from W[N,K] row-major (i.e. B column-major).
// ---------------------------------------------------------------------------
__device__ __forceinline__ v16h load_frag(const _Float16* p) {
  v8h lo = *(const v8h*)(p);        // 16B contiguous
  v8h hi = *(const v8h*)(p + 16);   // 16B contiguous, +32B
  v16h v;
#pragma unroll
  for (int i = 0; i < 8; ++i) { v[i] = lo[i]; v[i + 8] = hi[i]; }
  return v;
}

__device__ __forceinline__ v8f wmma_f16(v16h a, v16h b, v8f c) {
  // emits v_wmma_f32_16x16x32_f16
  return __builtin_amdgcn_wmma_f32_16x16x32_f16(false, a, false, b,
                                                (short)0, c, false, false);
}

__device__ __forceinline__ float sigmoid_f(float x) {
  return 1.0f / (1.0f + __expf(-x));
}
__device__ __forceinline__ float tanh_f(float x) {
  return 2.0f / (1.0f + __expf(-2.0f * x)) - 1.0f;
}

// ---------------------------------------------------------------------------
// f32 -> f16 conversion (weights), grid-stride
// ---------------------------------------------------------------------------
__global__ void __launch_bounds__(256)
f32_to_f16_kernel(const float* __restrict__ src, _Float16* __restrict__ dst,
                  long n) {
  long i = (long)blockIdx.x * blockDim.x + threadIdx.x;
  long stride = (long)gridDim.x * blockDim.x;
  for (; i < n; i += stride) dst[i] = (_Float16)src[i];
}

__global__ void __launch_bounds__(256)
bias_sum_kernel(const float* __restrict__ a, const float* __restrict__ b,
                float* __restrict__ o, int n) {
  int i = blockIdx.x * blockDim.x + threadIdx.x;
  if (i < n) o[i] = a[i] + b[i];
}

// ---------------------------------------------------------------------------
// Build x0 f16 [T,B,2E]: [0:E) = emb[word[b]] (broadcast over t), [E:2E) = emb[seq[t,b]]
// One block per (t,b) pair.
// ---------------------------------------------------------------------------
__global__ void __launch_bounds__(256)
build_x0_kernel(const int* __restrict__ word, const int* __restrict__ seq,
                const float* __restrict__ emb, _Float16* __restrict__ x0) {
  int tb = blockIdx.x;          // t*B + b
  int b  = tb % B_;
  int w  = word[b];
  int s  = seq[tb];
  const float* ew = emb + (long)w * E_;
  const float* es = emb + (long)s * E_;
  _Float16* dst = x0 + (long)tb * (2 * E_);
  for (int i = threadIdx.x; i < E_; i += 256) {
    dst[i]      = (_Float16)ew[i];
    dst[E_ + i] = (_Float16)es[i];
  }
}

// ---------------------------------------------------------------------------
// Generic WMMA GEMM: C[M,N] (f32) = A[M,K] (f16, lda) @ W[N,K]^T (f16, ldw) + bias[N]
// K is a compile-time template parameter (1024 or 2048 here) -> known trip
// counts, immediate-offset addressing.
// WG = 256 thr = 8 waves; WG tile = 64 rows x 256 cols
//   wave w: row half = w&1 (2 row tiles = 32 rows), col group = w>>1 (4 tiles = 64 cols)
// Ping-pong software pipeline: two disjoint fragment buffer sets, 2 chunks per
// loop iteration -> no register rotation movs, partial s_wait_loadcnt only.
// Requires M%64==0, N%256==0, K%64==0 (all shapes here comply).
// ---------------------------------------------------------------------------
template <int K>
__global__ void __launch_bounds__(256)
gemm_wmma_kernel(const _Float16* __restrict__ A, int lda,
                 const _Float16* __restrict__ W, int ldw,
                 const float* __restrict__ bias,
                 float* __restrict__ C, long ldc,
                 int M, int N) {
  const int lane = threadIdx.x & 31;
  const int wave = threadIdx.x >> 5;
  const int m0 = blockIdx.y * 64 + (wave & 1) * 32;
  const int n0 = blockIdx.x * 256 + (wave >> 1) * 64;
  if (m0 >= M || n0 >= N) return;  // uniform per wave

  const int r  = lane & 15;
  const int kh = (lane >> 4) << 3;  // 0 or 8

  const _Float16* arow0 = A + (long)(m0 + r) * lda + kh;
  const _Float16* arow1 = A + (long)(m0 + 16 + r) * lda + kh;
  const _Float16* wrow[4];
#pragma unroll
  for (int j = 0; j < 4; ++j) wrow[j] = W + (long)(n0 + j * 16 + r) * ldw + kh;

  v8f acc[2][4];
#pragma unroll
  for (int i = 0; i < 2; ++i)
#pragma unroll
    for (int j = 0; j < 4; ++j) acc[i][j] = v8f{};

  v16h aP[2], bP[4];   // ping buffers
  v16h aQ[2], bQ[4];   // pong buffers

  auto load_set = [&](v16h (&a)[2], v16h (&b)[4], int k) {
    a[0] = load_frag(arow0 + k);
    a[1] = load_frag(arow1 + k);
#pragma unroll
    for (int j = 0; j < 4; ++j) b[j] = load_frag(wrow[j] + k);
  };
  auto mma_set = [&](v16h (&a)[2], v16h (&b)[4]) {
#pragma unroll
    for (int j = 0; j < 4; ++j) {
      acc[0][j] = wmma_f16(a[0], b[j], acc[0][j]);
      acc[1][j] = wmma_f16(a[1], b[j], acc[1][j]);
    }
  };

  constexpr int NC = K / 32;  // chunk count (even)
  load_set(aP, bP, 0);
  for (int c = 0; c < NC - 2; c += 2) {
    load_set(aQ, bQ, (c + 1) * 32);
    mma_set(aP, bP);
    load_set(aP, bP, (c + 2) * 32);
    mma_set(aQ, bQ);
  }
  load_set(aQ, bQ, (NC - 1) * 32);
  mma_set(aP, bP);
  mma_set(aQ, bQ);

  // store with bias (D fragment: VGPR g -> M = g + 8*(lane>=16), N = lane&15)
  const int nlane = lane & 15;
  const int mg    = (lane >> 4) * 8;
#pragma unroll
  for (int i = 0; i < 2; ++i) {
#pragma unroll
    for (int j = 0; j < 4; ++j) {
      int n = n0 + j * 16 + nlane;
      float bv = bias ? bias[n] : 0.0f;
#pragma unroll
      for (int g = 0; g < 8; ++g) {
        int m = m0 + i * 16 + mg + g;
        C[(long)m * ldc + n] = acc[i][j][g] + bv;
      }
    }
  }
}

// ---------------------------------------------------------------------------
// Device-wide barrier (sense/generation). Counter+gen live in workspace,
// zeroed with hipMemsetAsync before the persistent kernels run.
// ---------------------------------------------------------------------------
__device__ __forceinline__ void grid_barrier(unsigned* cnt, unsigned* gen,
                                             unsigned nwg) {
  __syncthreads();
  if (threadIdx.x == 0) {
    unsigned g = __hip_atomic_load(gen, __ATOMIC_ACQUIRE,
                                   __HIP_MEMORY_SCOPE_AGENT);
    unsigned a = __hip_atomic_fetch_add(cnt, 1u, __ATOMIC_ACQ_REL,
                                        __HIP_MEMORY_SCOPE_AGENT);
    if (a == nwg - 1) {
      __hip_atomic_store(cnt, 0u, __ATOMIC_RELAXED, __HIP_MEMORY_SCOPE_AGENT);
      __hip_atomic_fetch_add(gen, 1u, __ATOMIC_ACQ_REL,
                             __HIP_MEMORY_SCOPE_AGENT);
    } else {
      while (__hip_atomic_load(gen, __ATOMIC_ACQUIRE,
                               __HIP_MEMORY_SCOPE_AGENT) == g) {
        __builtin_amdgcn_s_sleep(1);
      }
    }
  }
  __syncthreads();
}

// ---------------------------------------------------------------------------
// Persistent LSTM layer: 50 steps inside one kernel, grid-wide sync per step.
// Grid: 64 WGs x 256 thr. WG owns a 16-wide column slice n0 of H; computes all
// 4 gate blocks for all 64 batch rows there:
//   wave w: row tile rt=w&3 (16 of 64 rows), gate pair gp=w>>2 (gates 2gp,2gp+1)
// Gates are combined through LDS; cell state c stays in registers (fixed
// thread->cell ownership across all timesteps). K-loop ping-pong pipelined.
// ---------------------------------------------------------------------------
__global__ void __launch_bounds__(256)
lstm_layer_kernel(const _Float16* __restrict__ Whh,   // [4H,H] f16
                  const float* __restrict__ xg,       // [T,B,4H] f32 (bias folded)
                  const float* __restrict__ c0,       // [B,H] f32
                  _Float16* __restrict__ h16,         // [B,H] f16 (zeroed)
                  _Float16* __restrict__ hs,          // [T,B,H] f16 out
                  float* __restrict__ h_final,        // [B,H] f32 out
                  float* __restrict__ c_final,        // [B,H] f32 out
                  unsigned* __restrict__ bar,         // [2] cnt, gen
                  int nsteps, unsigned nwg) {
  __shared__ float lds_g[4][B_][16];   // 16 KB

  const int tid  = threadIdx.x;
  const int lane = tid & 31;
  const int wave = tid >> 5;
  const int n0   = blockIdx.x * 16;    // column slice of H
  const int rt   = wave & 3;           // row tile (16 rows of B)
  const int gp   = wave >> 2;          // gate pair: gates 2gp, 2gp+1

  const int r  = lane & 15;
  const int kh = (lane >> 4) << 3;

  // Per-thread cell-state registers: cells (tid + 256*j), m=cell>>4, nc=cell&15
  float creg[4];
#pragma unroll
  for (int j = 0; j < 4; ++j) {
    int cell = tid + 256 * j;
    creg[j] = c0[(long)(cell >> 4) * H_ + n0 + (cell & 15)];
  }

  const _Float16* w0 = Whh + (long)((2 * gp)     * H_ + n0 + r) * H_ + kh;
  const _Float16* w1 = Whh + (long)((2 * gp + 1) * H_ + n0 + r) * H_ + kh;
  const _Float16* ar = h16 + (long)(rt * 16 + r) * H_ + kh;

  for (int t = 0; t < nsteps; ++t) {
    // --- gates[B,16] for 2 gates: h @ Whh^T slice, K = H (ping-pong) ---
    v8f acc0 = v8f{}, acc1 = v8f{};
    v16h aP, b0P, b1P, aQ, b0Q, b1Q;

    auto load_set = [&](v16h& a, v16h& b0, v16h& b1, int k) {
      a  = load_frag(ar + k);
      b0 = load_frag(w0 + k);
      b1 = load_frag(w1 + k);
    };
    auto mma_set = [&](v16h& a, v16h& b0, v16h& b1) {
      acc0 = wmma_f16(a, b0, acc0);
      acc1 = wmma_f16(a, b1, acc1);
    };

    constexpr int NC = H_ / 32;  // 32 chunks (even)
    load_set(aP, b0P, b1P, 0);
    for (int c = 0; c < NC - 2; c += 2) {
      load_set(aQ, b0Q, b1Q, (c + 1) * 32);
      mma_set(aP, b0P, b1P);
      load_set(aP, b0P, b1P, (c + 2) * 32);
      mma_set(aQ, b0Q, b1Q);
    }
    load_set(aQ, b0Q, b1Q, (NC - 1) * 32);
    mma_set(aP, b0P, b1P);
    mma_set(aQ, b0Q, b1Q);

    // --- add xg[t] and stage in LDS ---
    const float* xgt = xg + (long)t * (B_ * H4_);
    const int mb = rt * 16 + (lane >> 4) * 8;
    const int nc = lane & 15;
#pragma unroll
    for (int g = 0; g < 8; ++g) {
      int m = mb + g;
      lds_g[2 * gp][m][nc] =
          acc0[g] + xgt[(long)m * H4_ + (2 * gp) * H_ + n0 + nc];
      lds_g[2 * gp + 1][m][nc] =
          acc1[g] + xgt[(long)m * H4_ + (2 * gp + 1) * H_ + n0 + nc];
    }
    __syncthreads();
    // --- elementwise LSTM cell update (i,f,g,o), c in registers ---
#pragma unroll
    for (int j = 0; j < 4; ++j) {
      int cell = tid + 256 * j;
      int m = cell >> 4, ncc = cell & 15;
      float iv = lds_g[0][m][ncc];
      float fv = lds_g[1][m][ncc];
      float gv = lds_g[2][m][ncc];
      float ov = lds_g[3][m][ncc];
      float cn = sigmoid_f(fv) * creg[j] + sigmoid_f(iv) * tanh_f(gv);
      float hn = sigmoid_f(ov) * tanh_f(cn);
      creg[j] = cn;
      long idx = (long)m * H_ + n0 + ncc;
      _Float16 h16v = (_Float16)hn;
      h16[idx] = h16v;
      hs[(long)t * (B_ * H_) + idx] = h16v;
      if (t == nsteps - 1) {
        h_final[idx] = hn;
        c_final[idx] = cn;
      }
    }
    __syncthreads();                 // protect LDS reuse
    grid_barrier(bar, bar + 1, nwg); // publish h16 device-wide
  }
}

// ---------------------------------------------------------------------------
// Host side
// ---------------------------------------------------------------------------
extern "C" void kernel_launch(void* const* d_in, const int* in_sizes, int n_in,
                              void* d_out, int out_size, void* d_ws,
                              size_t ws_size, hipStream_t stream) {
  (void)in_sizes; (void)n_in; (void)out_size; (void)ws_size;

  const int*   word = (const int*)  d_in[0];
  const int*   seq  = (const int*)  d_in[1];
  const float* emb  = (const float*)d_in[2];
  const float* w2hW = (const float*)d_in[3];
  const float* w2hb = (const float*)d_in[4];
  const float* Wih0 = (const float*)d_in[5];
  const float* Whh0 = (const float*)d_in[6];
  const float* bih0 = (const float*)d_in[7];
  const float* bhh0 = (const float*)d_in[8];
  const float* Wih1 = (const float*)d_in[9];
  const float* Whh1 = (const float*)d_in[10];
  const float* bih1 = (const float*)d_in[11];
  const float* bhh1 = (const float*)d_in[12];
  const float* decW = (const float*)d_in[13];
  const float* decb = (const float*)d_in[14];
  float* out = (float*)d_out;

  // Workspace carve-up (256B aligned blocks)
  char* ws = (char*)d_ws;
  size_t off = 0;
  auto carve = [&](size_t bytes) -> void* {
    void* p = ws + off;
    off += (bytes + 255) & ~(size_t)255;
    return p;
  };
  _Float16* x0     = (_Float16*)carve((size_t)T_ * B_ * 2 * E_ * 2);   // 13.1 MB
  _Float16* w2h16  = (_Float16*)carve((size_t)H_ * E_ * 2);            //  2.1 MB
  _Float16* wih016 = (_Float16*)carve((size_t)H4_ * 2 * E_ * 2);       // 16.8 MB
  _Float16* whh016 = (_Float16*)carve((size_t)H4_ * H_ * 2);           //  8.4 MB
  _Float16* wih116 = (_Float16*)carve((size_t)H4_ * H_ * 2);           //  8.4 MB
  _Float16* whh116 = (_Float16*)carve((size_t)H4_ * H_ * 2);           //  8.4 MB
  _Float16* dec16  = (_Float16*)carve((size_t)V_ * H_ * 2);            // 65.5 MB
  float*    bsum0  = (float*)   carve((size_t)H4_ * 4);
  float*    bsum1  = (float*)   carve((size_t)H4_ * 4);
  float*    c0buf  = (float*)   carve((size_t)B_ * H_ * 4);
  float*    xgbuf  = (float*)   carve((size_t)T_ * B_ * H4_ * 4);      // 52.4 MB (reused)
  _Float16* hs0    = (_Float16*)carve((size_t)T_ * B_ * H_ * 2);       //  6.6 MB
  _Float16* hs1    = (_Float16*)carve((size_t)T_ * B_ * H_ * 2);       //  6.6 MB
  _Float16* h16    = (_Float16*)carve((size_t)B_ * H_ * 2);            //  128 KB
  unsigned* bar    = (unsigned*)carve(256);

  // Output sub-buffers: decoded [T*B,V], then h_final [2,B,H], c_final [2,B,H]
  float* hfin0 = out + (size_t)T_ * B_ * V_;
  float* hfin1 = hfin0 + (size_t)B_ * H_;
  float* cfin0 = hfin0 + (size_t)2 * B_ * H_;
  float* cfin1 = cfin0 + (size_t)B_ * H_;

  // 0. reset barrier state (deterministic per call; graph-capturable)
  hipMemsetAsync(bar, 0, 256, stream);

  // 1. weight conversion f32 -> f16
  auto cvt = [&](const float* s, _Float16* d, long n) {
    f32_to_f16_kernel<<<4096, 256, 0, stream>>>(s, d, n);
  };
  cvt(w2hW, w2h16, (long)H_ * E_);
  cvt(Wih0, wih016, (long)H4_ * 2 * E_);
  cvt(Whh0, whh016, (long)H4_ * H_);
  cvt(Wih1, wih116, (long)H4_ * H_);
  cvt(Whh1, whh116, (long)H4_ * H_);
  cvt(decW, dec16, (long)V_ * H_);

  // 2. combined biases (b_ih + b_hh folded into xg GEMM)
  bias_sum_kernel<<<(H4_ + 255) / 256, 256, 0, stream>>>(bih0, bhh0, bsum0, H4_);
  bias_sum_kernel<<<(H4_ + 255) / 256, 256, 0, stream>>>(bih1, bhh1, bsum1, H4_);

  // 3. embedding gather + concat -> x0 f16 [T,B,2E]
  build_x0_kernel<<<T_ * B_, 256, 0, stream>>>(word, seq, emb, x0);

  // 4. c0 = word_emb @ w2h_W^T + w2h_b    [64,1024], K=1024
  //    word_emb = first half of x0[t=0] rows (lda = 2E)
  gemm_wmma_kernel<E_><<<dim3(H_ / 256, B_ / 64), 256, 0, stream>>>(
      x0, 2 * E_, w2h16, E_, w2hb, c0buf, H_, B_, H_);

  // 5. xg0 = x0 @ W_ih0^T + bsum0         [3200,4096], K=2048
  gemm_wmma_kernel<2 * E_><<<dim3(H4_ / 256, (T_ * B_) / 64), 256, 0, stream>>>(
      x0, 2 * E_, wih016, 2 * E_, bsum0, xgbuf, H4_, T_ * B_, H4_);

  // 6. layer-0 recurrence (persistent, 64 WGs = H/16 column slices)
  hipMemsetAsync(h16, 0, (size_t)B_ * H_ * 2, stream);   // h0 = 0
  lstm_layer_kernel<<<H_ / 16, 256, 0, stream>>>(
      whh016, xgbuf, c0buf, h16, hs0, hfin0, cfin0, bar, T_, H_ / 16);

  // 7. xg1 = hs0 @ W_ih1^T + bsum1        [3200,4096], K=1024 (reuses xgbuf)
  gemm_wmma_kernel<H_><<<dim3(H4_ / 256, (T_ * B_) / 64), 256, 0, stream>>>(
      hs0, H_, wih116, H_, bsum1, xgbuf, H4_, T_ * B_, H4_);

  // 8. layer-1 recurrence (same h0=0, c0 per reference)
  hipMemsetAsync(h16, 0, (size_t)B_ * H_ * 2, stream);
  lstm_layer_kernel<<<H_ / 16, 256, 0, stream>>>(
      whh116, xgbuf, c0buf, h16, hs1, hfin1, cfin1, bar, T_, H_ / 16);

  // 9. decoder: out = hs1 @ dec_W^T + dec_b   [3200,32000], K=1024
  gemm_wmma_kernel<H_><<<dim3(V_ / 256, (T_ * B_) / 64), 256, 0, stream>>>(
      hs1, H_, dec16, H_, decb, out, V_, T_ * B_, V_);
}
